// SelfAttentionLayer_20916490731687
// MI455X (gfx1250) — matrix operand order repaired and checked by reading, not compiled, not run on hardware
//
#include <hip/hip_runtime.h>
#include <type_traits>

typedef __attribute__((ext_vector_type(16))) _Float16 v16h;
typedef __attribute__((ext_vector_type(8)))  _Float16 v8h;
typedef __attribute__((ext_vector_type(4)))  _Float16 v4h;
typedef __attribute__((ext_vector_type(8)))  float    v8f;

#define NB  4
#define NT  2048
#define NC  1024
#define NH  16
#define NHD 64
#define LDSPAD 40   // halves per staged B-tile row (32 data + 8 pad -> 80B rows)

__device__ __forceinline__ v16h cat8(v8h lo, v8h hi) {
  return __builtin_shufflevector(lo, hi, 0,1,2,3,4,5,6,7,8,9,10,11,12,13,14,15);
}

__device__ __forceinline__ v8f wmma32(v16h a, v16h b, v8f c) {
  // D = A(16x32 f16) * B(32x16 f16) + C(16x16 f32)
  return __builtin_amdgcn_wmma_f32_16x16x32_f16(false, a, false, b, (short)0, c,
                                                false, false);
}

// raw v_exp_f32 (2^x) + v_med3 clamp: p in [0, 32768], no libm denorm fixups
__device__ __forceinline__ float exp2_clamped(float x) {
  return __builtin_amdgcn_fmed3f(__builtin_amdgcn_exp2f(x), 0.f, 32768.f);
}

// ds_swizzle_b32 lane exchange (imm-encoded pattern, no address VGPR)
template<int IMM>
__device__ __forceinline__ float swz(float x) {
  return __int_as_float(__builtin_amdgcn_ds_swizzle(__float_as_int(x), IMM));
}
#define SWZ_BC16  0x0010   // src = lane & 0x10 (broadcast lane 0 / 16 per half)

// async 16B copy global -> LDS (ASYNCcnt-tracked, CDNA5)
__device__ __forceinline__ void async_cp16(unsigned lds_off, const void* gptr) {
  asm volatile("global_load_async_to_lds_b128 %0, %1, off"
               :: "v"(lds_off), "v"((unsigned long long)(uintptr_t)gptr)
               : "memory");
}

// ---------------- f32 -> f16 convert (x) ----------------
__global__ __launch_bounds__(256)
void cvt_f16(const float* __restrict__ in, _Float16* __restrict__ out) {
  size_t i = ((size_t)blockIdx.x * 256 + threadIdx.x) * 4;
  const float4 v = *(const float4*)(in + i);
  v4h o;
  o[0] = (_Float16)v.x; o[1] = (_Float16)v.y;
  o[2] = (_Float16)v.z; o[3] = (_Float16)v.w;
  *(v4h*)(out + i) = o;
}

// ---------------- W (KxN f32) -> W^T (NxK f16) ----------------
__global__ __launch_bounds__(256)
void transpose_cvt(const float* __restrict__ W, _Float16* __restrict__ WT) {
  __shared__ _Float16 tile[32][33];
  const int lx = threadIdx.x & 31;
  const int wy = threadIdx.x >> 5;           // 0..7
  const int gx = blockIdx.x * 32 + lx;       // N index
  const int gy = blockIdx.y * 32 + wy * 4;   // K index
#pragma unroll
  for (int i = 0; i < 4; ++i)
    tile[wy * 4 + i][lx] = (_Float16)W[(size_t)(gy + i) * NC + gx];
  __syncthreads();
  const int ox = blockIdx.y * 32 + lx;       // K index
  const int oy = blockIdx.x * 32 + wy * 4;   // N index
#pragma unroll
  for (int i = 0; i < 4; ++i)
    WT[(size_t)(oy + i) * NC + ox] = tile[lx][wy * 4 + i];
}

// ---------------- WMMA GEMM with async-LDS double-buffered B tile ----------
// out = A(8192x1024 f16) @ BT^T + bias
// MODE 0: f16 out, layout (B,H,T,HD)                    (K)
// MODE 1: f16 out, layout (B,H,HD,T)                    (V transposed)
// MODE 2: f32 out, row-major (B*T, C)                   (final projection)
// MODE 3: f16 out, (B,H,T,HD), * log2(e)/sqrt(T)        (Q, pre-scaled)
template<int MODE>
__global__ __launch_bounds__(256)
void gemm16(const _Float16* __restrict__ A, const _Float16* __restrict__ BT,
            const float* __restrict__ bias, void* __restrict__ outp)
{
  __shared__ __align__(16) _Float16 btile[2][64 * LDSPAD];
  const int tid  = threadIdx.x;
  const int lane = tid & 31;
  const int wave = tid >> 5;
  const int n15  = lane & 15;
  const int half = lane >> 4;
  const int kbA  = half * 8;    // A fragment K sub-offset
  const int kbB  = half * 16;   // unused here (B fragment also uses kbA)

  const int m0 = blockIdx.y * 128 + wave * 16;
  const int n0 = blockIdx.x * 64;

  // cooperative staging: thread t copies 16B of the 64x32 B tile
  const int brow = tid >> 2;          // 0..63  (tile row = output column)
  const int bcol = (tid & 3) * 8;     // 0,8,16,24 (halves)
  const _Float16* Bsrc = BT + (size_t)(n0 + brow) * NC + bcol;
  const unsigned ldsdst[2] = {
      (unsigned)(uintptr_t)&btile[0][brow * LDSPAD + bcol],
      (unsigned)(uintptr_t)&btile[1][brow * LDSPAD + bcol]};

  const _Float16* Arow = A + (size_t)(m0 + n15) * NC;

  async_cp16(ldsdst[0], Bsrc);        // prefetch first tile
  v8f acc[4] = {};
  int cur = 0;
  for (int k = 0; k < NC; k += 32) {
    if (k + 32 < NC) {
      async_cp16(ldsdst[cur ^ 1], Bsrc + (k + 32));     // overlap next tile
      asm volatile("s_wait_asynccnt 0x1" ::: "memory"); // current tile landed
    } else {
      asm volatile("s_wait_asynccnt 0x0" ::: "memory");
    }
    __syncthreads();                  // all waves' contributions visible

    v16h af = cat8(*(const v8h*)(Arow + k + kbA),
                   *(const v8h*)(Arow + k + kbA + 16));
    const _Float16* Lb = &btile[cur][0];
#pragma unroll
    for (int j = 0; j < 4; ++j) {
      const _Float16* Bp = Lb + (j * 16 + n15) * LDSPAD + kbA;
      v16h bf = cat8(*(const v8h*)Bp, *(const v8h*)(Bp + 16));
      acc[j] = wmma32(af, bf, acc[j]);
    }
    __syncthreads();                  // protect buffer before next overwrite
    cur ^= 1;
  }

  // logits pre-scale: 1/sqrt(T) * log2(e)  (softmax later runs in base 2)
  const float SCL = 0.022097086912079608f * 1.4426950408889634f;
#pragma unroll
  for (int j = 0; j < 4; ++j) {
    const int col = n0 + j * 16 + n15;
    const float bv = bias[col];
#pragma unroll
    for (int r = 0; r < 8; ++r) {
      const int row = m0 + r + 8 * half;
      float val = acc[j][r] + bv;
      if (MODE == 3) val *= SCL;
      if (MODE == 2) {
        ((float*)outp)[(size_t)row * NC + col] = val;
      } else {
        const int b = row >> 11;   // / NT
        const int t = row & 2047;
        const int h = col >> 6;    // / NHD
        const int d = col & 63;
        if (MODE == 1)
          ((_Float16*)outp)[((size_t)(b * NH + h) * NHD + d) * NT + t] = (_Float16)val;
        else
          ((_Float16*)outp)[((size_t)(b * NH + h) * NT + t) * NHD + d] = (_Float16)val;
      }
    }
  }
}

// ---------------- flash attention: one wave per 16-query tile ----------------
// Softmax is shift-invariant; logits here are (q.k)*log2(e)/sqrt(2048) = O(0.1),
// so no running-max is needed. P is clamped to [0, 2^15] after exp2, so the
// f16-staged P and the f32 row sum (<= 2048 * 2^15 = 2^26) can never overflow.
__global__ __launch_bounds__(256)
void attn(const _Float16* __restrict__ Q, const _Float16* __restrict__ Km,
          const _Float16* __restrict__ VT, _Float16* __restrict__ O)
{
  __shared__ __align__(16) _Float16 lds[8][16 * 32];
  const int lane  = threadIdx.x & 31;
  const int wave  = threadIdx.x >> 5;
  const int gid   = blockIdx.x * 8 + wave;   // 0..8191
  const int qtile = gid & 127;               // T/16
  const int bh    = gid >> 7;                // 0..63
  const int bb    = bh >> 4;
  const int hh    = bh & 15;

  const int q0   = qtile * 16;
  const int n15  = lane & 15;
  const int half = lane >> 4;
  const int kbA  = half * 8;
  const int kbB  = half * 16;

  // Q carries log2(e)/sqrt(T); v_exp_f32 is natively 2^x
  const _Float16* Qrow = Q + ((size_t)bh * NT + q0 + n15) * NHD;
  v16h aq0 = cat8(*(const v8h*)(Qrow + kbB),      *(const v8h*)(Qrow + kbB + 8));
  v16h aq1 = cat8(*(const v8h*)(Qrow + 32 + kbB), *(const v8h*)(Qrow + 32 + kbB + 8));

  // "ones column" B fragment: column 0 all ones -> accL accumulates row sums
  v16h bones;
#pragma unroll
  for (int i = 0; i < 16; ++i) bones[i] = (n15 == 0) ? (_Float16)1.f : (_Float16)0.f;

  v8f acc[4] = {};
  v8f accL = {};                     // softmax denominator accumulator

  _Float16* Lp = &lds[wave][0];

  auto chunk = [&](int key0, auto MASKED) {
    const _Float16* KrowL = Km + ((size_t)bh * NT + key0 + n15) * NHD;
    const _Float16* KrowR = KrowL + 16 * NHD;
    v8f sL = {};
    sL = wmma32(aq0, *(const v16h*)(KrowL + kbB), sL);
    sL = wmma32(aq1, *(const v16h*)(KrowL + 32 + kbB), sL);
    v8f sR = {};
    sR = wmma32(aq0, *(const v16h*)(KrowR + kbB), sR);
    sR = wmma32(aq1, *(const v16h*)(KrowR + 32 + kbB), sR);

    const int keyL = key0 + n15;
    const int keyR = keyL + 16;
#pragma unroll
    for (int r = 0; r < 8; ++r) {
      float sl = sL[r];
      float sr = sR[r];
      if constexpr (decltype(MASKED)::value) {
        const int qi = q0 + r + 8 * half;
        sl = (keyL <= qi) ? sl : -1e30f;       // exp2 -> 0
        sr = (keyR <= qi) ? sr : -1e30f;
      }
      const float pl = exp2_clamped(sl);       // v_exp_f32 + v_med3_f32
      const float pr = exp2_clamped(sr);
      const auto pk = __builtin_amdgcn_cvt_pkrtz(pl, pr);  // one packed cvt
      const int row = r + 8 * half;
      Lp[row * 32 + n15]      = (_Float16)pk[0];  // stage P in A-layout
      Lp[row * 32 + 16 + n15] = (_Float16)pk[1];
    }
    asm volatile("s_wait_dscnt 0x0" ::: "memory");    // wave-synchronous relayout
    v16h pf = cat8(*(const v8h*)(Lp + n15 * 32 + kbB),
                   *(const v8h*)(Lp + n15 * 32 + kbB + 8));
    accL = wmma32(pf, bones, accL);            // denominator += rowsum(P)
    // O += P (16x32) @ V (32x64); V stored transposed -> contiguous B frags
#pragma unroll
    for (int j = 0; j < 4; ++j) {
      const _Float16* Vp = VT + ((size_t)bh * NHD + j * 16 + n15) * NT + key0 + kbB;
      acc[j] = wmma32(pf, *(const v16h*)Vp, acc[j]);
    }
  };

  // causal split: q0 is a multiple of 16 -> exactly one diagonal (masked) chunk
  const int nfull = q0 >> 5;                  // fully-unmasked 32-key chunks
  for (int c = 0; c < nfull; ++c)
    chunk(c * 32, std::false_type{});
  chunk(nfull * 32, std::true_type{});        // boundary chunk with causal mask

#pragma unroll
  for (int r = 0; r < 8; ++r) {
    const float ls  = swz<SWZ_BC16>(accL[r]); // broadcast denom from col-0 lane
    const float inv = 1.0f / ls;
    const int t = q0 + r + 8 * half;
    _Float16* Op = O + ((size_t)bb * NT + t) * NC + hh * NHD;
#pragma unroll
    for (int j = 0; j < 4; ++j)
      Op[j * 16 + n15] = (_Float16)(acc[j][r] * inv);
  }
}

extern "C" void kernel_launch(void* const* d_in, const int* in_sizes, int n_in,
                              void* d_out, int out_size, void* d_ws, size_t ws_size,
                              hipStream_t stream) {
  const float* x  = (const float*)d_in[0];
  const float* Wq = (const float*)d_in[1];
  const float* Wk = (const float*)d_in[2];
  const float* Wv = (const float*)d_in[3];
  const float* Wp = (const float*)d_in[4];
  const float* bq = (const float*)d_in[5];
  const float* bk = (const float*)d_in[6];
  const float* bv = (const float*)d_in[7];
  const float* bp = (const float*)d_in[8];

  char* ws = (char*)d_ws;
  const size_t szX = (size_t)NB * NT * NC * sizeof(_Float16); // 16 MB
  const size_t szW = (size_t)NC * NC * sizeof(_Float16);      // 2 MB
  _Float16* x16 = (_Float16*)ws; ws += szX;
  _Float16* WqT = (_Float16*)ws; ws += szW;
  _Float16* WkT = (_Float16*)ws; ws += szW;
  _Float16* WvT = (_Float16*)ws; ws += szW;
  _Float16* WpT = (_Float16*)ws; ws += szW;
  _Float16* Qh  = (_Float16*)ws; ws += szX;   // (B,H,T,HD), pre-scaled
  _Float16* Kh  = (_Float16*)ws; ws += szX;   // (B,H,T,HD)
  _Float16* VTh = (_Float16*)ws; ws += szX;   // (B,H,HD,T)
  _Float16* Oat = (_Float16*)ws; ws += szX;   // (B,T,C)

  cvt_f16<<<(NB * NT * NC) / 1024, 256, 0, stream>>>(x, x16);
  dim3 tg(NC / 32, NC / 32);
  transpose_cvt<<<tg, 256, 0, stream>>>(Wq, WqT);
  transpose_cvt<<<tg, 256, 0, stream>>>(Wk, WkT);
  transpose_cvt<<<tg, 256, 0, stream>>>(Wv, WvT);
  transpose_cvt<<<tg, 256, 0, stream>>>(Wp, WpT);

  dim3 gg(NC / 64, (NB * NT) / 128);
  gemm16<3><<<gg, 256, 0, stream>>>(x16, WqT, bq, Qh);   // Q, scaled
  gemm16<0><<<gg, 256, 0, stream>>>(x16, WkT, bk, Kh);   // K
  gemm16<1><<<gg, 256, 0, stream>>>(x16, WvT, bv, VTh);  // V^T

  attn<<<(NB * NH * (NT / 16)) / 8, 256, 0, stream>>>(Qh, Kh, VTh, Oat);

  gemm16<2><<<gg, 256, 0, stream>>>(Oat, WpT, bp, d_out);
}
